// FixedRelPosAttn_30588757082923
// MI455X (gfx1250) — matrix operand order repaired and checked by reading, not compiled, not run on hardware
//
#include <hip/hip_runtime.h>

// Problem constants (match reference)
#define B_  8
#define N_  1024
#define D_  512
#define H_  8
#define DK_ 64
#define P_  2047

typedef __attribute__((ext_vector_type(16))) _Float16 v16h;
typedef __attribute__((ext_vector_type(8)))  _Float16 v8h;
typedef __attribute__((ext_vector_type(8)))  float    v8f;

static __device__ __forceinline__ v8f wmma_f16(v16h a, v16h b, v8f c) {
  // D = A(16x32) * B(32x16) + C, fp32 accumulate
  return __builtin_amdgcn_wmma_f32_16x16x32_f16(false, a, false, b, (short)0, c, false, false);
}

// A-matrix fragment (16x32 f16). Per ISA layout, lane holds K = {kb..kb+7} and
// {kb+16..kb+23} with kb = (lane>=16)*8 -> two contiguous 16B chunks.
static __device__ __forceinline__ v16h ldA16(const _Float16* rowbase, int lane, int koff) {
  const int kb = (lane >> 4) * 8;
  v8h lo = *(const v8h*)(rowbase + koff + kb);
  v8h hi = *(const v8h*)(rowbase + koff + kb + 16);
  return __builtin_shufflevector(lo, hi, 0,1,2,3,4,5,6,7,8,9,10,11,12,13,14,15);
}

// B-matrix fragment (32x16 f16). Lane holds K = {kb..kb+7} and {kb+16..kb+23},
// kb=(lane>=16)*8, when source is stored [n][K] (transposed).
static __device__ __forceinline__ v16h ldB16(const _Float16* rowbase, int lane, int koff) {
  const int kb = (lane >> 4) * 8;
  v8h lo = *(const v8h*)(rowbase + koff + kb);
  v8h hi = *(const v8h*)(rowbase + koff + kb + 16);
  return __builtin_shufflevector(lo, hi, 0,1,2,3,4,5,6,7,8,9,10,11,12,13,14,15);
}

// A-matrix fragment from fp32 source (scores in LDS), converting to f16.
static __device__ __forceinline__ v16h ldA16_f32(const float* rowbase, int lane, int koff) {
  const int kb = (lane >> 4) * 8;
  const float4* p0 = (const float4*)(rowbase + koff + kb);
  const float4* p1 = (const float4*)(rowbase + koff + kb + 16);
  float4 a0 = p0[0], a1 = p0[1];
  float4 b0 = p1[0], b1 = p1[1];
  v16h r;
  r[0]  = (_Float16)a0.x; r[1]  = (_Float16)a0.y; r[2]  = (_Float16)a0.z; r[3]  = (_Float16)a0.w;
  r[4]  = (_Float16)a1.x; r[5]  = (_Float16)a1.y; r[6]  = (_Float16)a1.z; r[7]  = (_Float16)a1.w;
  r[8]  = (_Float16)b0.x; r[9]  = (_Float16)b0.y; r[10] = (_Float16)b0.z; r[11] = (_Float16)b0.w;
  r[12] = (_Float16)b1.x; r[13] = (_Float16)b1.y; r[14] = (_Float16)b1.z; r[15] = (_Float16)b1.w;
  return r;
}

// ---------------------------------------------------------------------------
// Projection GEMM: Y[m,e] = sum_d X[m,d]*W[d,e] (+bias[e]) (+extra[e])
// X fp32 [M,512], W fp32 [512,512]; out f16:
//   vtrans=0: [B,H,L,DK] layout (e = h*64+dk)
//   vtrans=1: [B,H,DK,L] layout (transposed for attn@v B-fragments)
// Optionally writes a second output with a different extra bias (qu / qv).
// Block: 256 threads = 8 waves, tile 64(M) x 64(N), K-step 32.
// ---------------------------------------------------------------------------
__global__ __launch_bounds__(256) void rpa_proj_kernel(
    const float* __restrict__ X, const float* __restrict__ W,
    const float* __restrict__ bias,
    const float* __restrict__ extra1, _Float16* __restrict__ out1,
    const float* __restrict__ extra2, _Float16* __restrict__ out2,
    int M, int L, int vtrans)
{
  __shared__ _Float16 Xs[64][40];   // [row][k], row stride 80B (16B aligned)
  __shared__ _Float16 Wt[64][40];   // [col][k] transposed W tile

  const int tid  = threadIdx.x;
  const int lane = tid & 31;
  const int w    = tid >> 5;        // wave 0..7 -> 4(M) x 2(N)
  const int wm   = w >> 1;
  const int wn   = w & 1;
  const int mbase = blockIdx.x * 64;
  const int nbase = blockIdx.y * 64;

  v8f acc0 = {}; v8f acc1 = {};

  const int xr = tid >> 2;          // 0..63 (X tile row)
  const int xc = (tid & 3) * 8;     // 0,8,16,24 (X tile k)
  const int wr = tid >> 3;          // 0..31 (W tile k)
  const int wc = (tid & 7) * 8;     // 0..56 (W tile col)

  const _Float16* aRow  = &Xs[wm * 16 + (lane & 15)][0];
  const _Float16* bRow0 = &Wt[wn * 32 + (lane & 15)][0];
  const _Float16* bRow1 = &Wt[wn * 32 + 16 + (lane & 15)][0];

  for (int k0 = 0; k0 < D_; k0 += 32) {
    // Stage X tile (64x32) fp32->f16, vectorized
    {
      int gr = mbase + xr;
      float4 f0 = make_float4(0.f, 0.f, 0.f, 0.f), f1 = f0;
      if (gr < M) {
        const float* src = X + (size_t)gr * D_ + k0 + xc;
        f0 = *(const float4*)(src);
        f1 = *(const float4*)(src + 4);
        if (k0 + 32 < D_) __builtin_prefetch(src + 32, 0, 0);
      }
      v8h t;
      t[0] = (_Float16)f0.x; t[1] = (_Float16)f0.y; t[2] = (_Float16)f0.z; t[3] = (_Float16)f0.w;
      t[4] = (_Float16)f1.x; t[5] = (_Float16)f1.y; t[6] = (_Float16)f1.z; t[7] = (_Float16)f1.w;
      *(v8h*)&Xs[xr][xc] = t;
    }
    // Stage W tile (32x64) fp32->f16, transposed into Wt[col][k]
    {
      const float* src = W + (size_t)(k0 + wr) * D_ + nbase + wc;
      float4 f0 = *(const float4*)(src);
      float4 f1 = *(const float4*)(src + 4);
      if (k0 + 32 < D_) __builtin_prefetch(src + 32 * D_, 0, 0);
      Wt[wc + 0][wr] = (_Float16)f0.x; Wt[wc + 1][wr] = (_Float16)f0.y;
      Wt[wc + 2][wr] = (_Float16)f0.z; Wt[wc + 3][wr] = (_Float16)f0.w;
      Wt[wc + 4][wr] = (_Float16)f1.x; Wt[wc + 5][wr] = (_Float16)f1.y;
      Wt[wc + 6][wr] = (_Float16)f1.z; Wt[wc + 7][wr] = (_Float16)f1.w;
    }
    __syncthreads();
    {
      v16h a  = ldA16(aRow,  lane, 0);
      v16h b0 = ldB16(bRow0, lane, 0);
      v16h b1 = ldB16(bRow1, lane, 0);
      acc0 = wmma_f16(a, b0, acc0);
      acc1 = wmma_f16(a, b1, acc1);
    }
    __syncthreads();
  }

  // Epilogue: bias + extra, scatter f16
  const int rhi  = (lane >> 4) * 8;
  const int col0 = nbase + wn * 32 + (lane & 15);
  const int col1 = col0 + 16;
  const float bia0 = bias   ? bias[col0]   : 0.f;
  const float bia1 = bias   ? bias[col1]   : 0.f;
  const float e10  = extra1 ? extra1[col0] : 0.f;
  const float e11  = extra1 ? extra1[col1] : 0.f;
  const float e20  = extra2 ? extra2[col0] : 0.f;
  const float e21  = extra2 ? extra2[col1] : 0.f;
  #pragma unroll
  for (int r = 0; r < 8; ++r) {
    int m = mbase + wm * 16 + r + rhi;
    if (m >= M) continue;
    int bb = m / L;
    int ln = m - bb * L;
    size_t i0, i1;
    if (vtrans) {
      i0 = (((size_t)bb * H_ + (col0 >> 6)) * DK_ + (col0 & 63)) * (size_t)L + ln;
      i1 = (((size_t)bb * H_ + (col1 >> 6)) * DK_ + (col1 & 63)) * (size_t)L + ln;
    } else {
      i0 = (((size_t)bb * H_ + (col0 >> 6)) * (size_t)L + ln) * DK_ + (col0 & 63);
      i1 = (((size_t)bb * H_ + (col1 >> 6)) * (size_t)L + ln) * DK_ + (col1 & 63);
    }
    float y0 = acc0[r] + bia0;
    float y1 = acc1[r] + bia1;
    out1[i0] = (_Float16)(y0 + e10);
    out1[i1] = (_Float16)(y1 + e11);
    if (out2) {
      out2[i0] = (_Float16)(y0 + e20);
      out2[i1] = (_Float16)(y1 + e21);
    }
  }
}

// ---------------------------------------------------------------------------
// Fused relative-position attention: one block per (16-row tile, h, b).
// scores kept entirely in LDS; bd computed via 31-row sliding p-window.
// k/p are [B,H,N|P,DK] (K-contiguous for B-fragments); v is [B,H,DK,N].
// ---------------------------------------------------------------------------
__global__ __launch_bounds__(256) void rpa_attn_kernel(
    const _Float16* __restrict__ qu, const _Float16* __restrict__ qv,
    const _Float16* __restrict__ kh, const _Float16* __restrict__ vT,
    const _Float16* __restrict__ ph, _Float16* __restrict__ ao)
{
  __shared__ float    scores[16][1024];   // 64 KB
  __shared__ _Float16 qus[16][64];        // 2 KB
  __shared__ _Float16 qvs[16][64];        // 2 KB
  __shared__ float    bds[8][16][32];     // 16 KB (per-wave bd window; reused as partials)
  __shared__ float    red[16][16];        // 1 KB

  const int i0 = blockIdx.x * 16;
  const int h  = blockIdx.y;
  const int b  = blockIdx.z;
  const int tid  = threadIdx.x;
  const int lane = tid & 31;
  const int w    = tid >> 5;

  const size_t bh = (size_t)b * H_ + h;
  const _Float16* quP = qu + (bh * N_ + i0) * DK_;
  const _Float16* qvP = qv + (bh * N_ + i0) * DK_;
  const _Float16* kP  = kh + bh * (size_t)N_ * DK_;
  const _Float16* vP  = vT + bh * (size_t)DK_ * N_;   // transposed [DK][N]
  const _Float16* pP  = ph + bh * (size_t)P_ * DK_;

  // Stage qu/qv row tiles (16x64), 4 halves per thread
  {
    int x = tid * 4;                 // 0..1020
    int r = x >> 6, c = x & 63;
    *(uint2*)&qus[r][c] = *(const uint2*)(quP + r * DK_ + c);
    *(uint2*)&qvs[r][c] = *(const uint2*)(qvP + r * DK_ + c);
  }
  __syncthreads();

  const int col = lane & 15;
  const int rhi = (lane >> 4) * 8;
  const _Float16* aquRow = &qus[col][0];
  const _Float16* aqvRow = &qvs[col][0];

  // Phase 1: scores = (qu.kT + relshift(qv.pT)) / 8 ; 64 column tiles over 8 waves
  for (int itn = 0; itn < 8; ++itn) {
    const int j0 = (itn * 8 + w) * 16;
    const int w0 = (N_ - 1) - i0 - 15 + j0;   // first p-row of 31-row window (>= 0)
    const _Float16* kRow  = kP + (size_t)(j0 + col) * DK_;
    const _Float16* pRow0 = pP + (size_t)(w0 + col) * DK_;
    const _Float16* pRow1 = pP + (size_t)(w0 + 16 + col) * DK_;
    v8f ac = {}, bd0 = {}, bd1 = {};
    #pragma unroll
    for (int ks = 0; ks < 2; ++ks) {
      const int k32 = ks * 32;
      v16h aqu = ldA16(aquRow, lane, k32);
      v16h aqv = ldA16(aqvRow, lane, k32);
      v16h bk  = ldB16(kRow,  lane, k32);
      v16h bp0 = ldB16(pRow0, lane, k32);
      v16h bp1 = ldB16(pRow1, lane, k32);
      ac  = wmma_f16(aqu, bk,  ac);
      bd0 = wmma_f16(aqv, bp0, bd0);
      bd1 = wmma_f16(aqv, bp1, bd1);
    }
    // stash bd window (per-wave scratch), then gather along the shifted diagonal
    #pragma unroll
    for (int r = 0; r < 8; ++r) {
      bds[w][r + rhi][col]      = bd0[r];
      bds[w][r + rhi][col + 16] = bd1[r];
    }
    __builtin_amdgcn_wave_barrier();   // LDS in-order within wave; stop compiler reorder
    #pragma unroll
    for (int r = 0; r < 8; ++r) {
      int ii = r + rhi;
      float s = ac[r] + bds[w][ii][15 - ii + col];   // bd_full[i][N-1-i+j]
      scores[ii][j0 + col] = s * 0.125f;             // 1/sqrt(64)
    }
    __builtin_amdgcn_wave_barrier();
  }
  __syncthreads();

  // Phase 2: softmax over 1024 cols; 16 threads per row
  {
    const int r = tid >> 4, g = tid & 15;
    float mx = -3.0e38f;
    for (int c = 0; c < 64; ++c) mx = fmaxf(mx, scores[r][g * 64 + c]);
    red[r][g] = mx;
    __syncthreads();
    float rm = -3.0e38f;
    #pragma unroll
    for (int x = 0; x < 16; ++x) rm = fmaxf(rm, red[r][x]);
    __syncthreads();
    float sm = 0.f;
    for (int c = 0; c < 64; ++c) {
      float e = __expf(scores[r][g * 64 + c] - rm);
      scores[r][g * 64 + c] = e;
      sm += e;
    }
    red[r][g] = sm;
    __syncthreads();
    float rs = 0.f;
    #pragma unroll
    for (int x = 0; x < 16; ++x) rs += red[r][x];
    float inv = 1.0f / rs;
    for (int c = 0; c < 64; ++c) scores[r][g * 64 + c] *= inv;
  }
  __syncthreads();

  // Phase 3: out(16x64) = attn(16x1024) @ v(1024x64); 4 col-tiles x 2 K-halves
  {
    const int ct    = w & 3;    // dk column tile
    const int khalf = w >> 2;   // K half (512 each)
    const float*    sRow = &scores[col][0];
    const _Float16* vRow = vP + (size_t)(ct * 16 + col) * N_;  // [dk][j] contiguous in j
    v8f acc = {};
    for (int kt = 0; kt < 16; ++kt) {
      const int kg = khalf * 512 + kt * 32;
      v16h a  = ldA16_f32(sRow, lane, kg);
      v16h bb = ldB16(vRow, lane, kg);
      acc = wmma_f16(a, bb, acc);
    }
    float* part = &bds[0][0][0];   // reuse as [4][16][16] partial buffer
    if (khalf == 1) {
      #pragma unroll
      for (int r = 0; r < 8; ++r)
        part[(ct * 16 + r + rhi) * 16 + col] = acc[r];
    }
    __syncthreads();
    if (khalf == 0) {
      #pragma unroll
      for (int r = 0; r < 8; ++r) {
        float o = acc[r] + part[(ct * 16 + r + rhi) * 16 + col];
        int ii = r + rhi;
        // ao layout: [B, N, H*DK] == rows of the final GEMM
        ao[((size_t)b * N_ + i0 + ii) * D_ + h * DK_ + ct * 16 + col] = (_Float16)o;
      }
    }
  }
}

// ---------------------------------------------------------------------------
// Final GEMM: Y[m,e] = sum_d A[m,d]*Wout[d,e] + bout[e], A f16 [8192,512], Y fp32
// ---------------------------------------------------------------------------
__global__ __launch_bounds__(256) void rpa_final_kernel(
    const _Float16* __restrict__ A, const float* __restrict__ W,
    const float* __restrict__ bias, float* __restrict__ Y)
{
  __shared__ _Float16 Xs[64][40];
  __shared__ _Float16 Wt[64][40];   // transposed W tile [col][k]

  const int tid  = threadIdx.x;
  const int lane = tid & 31;
  const int w    = tid >> 5;
  const int wm   = w >> 1;
  const int wn   = w & 1;
  const int mbase = blockIdx.x * 64;
  const int nbase = blockIdx.y * 64;

  v8f acc0 = {}; v8f acc1 = {};

  const int xr = tid >> 2;
  const int xc = (tid & 3) * 8;
  const int wr = tid >> 3;
  const int wc = (tid & 7) * 8;

  const _Float16* aRow  = &Xs[wm * 16 + (lane & 15)][0];
  const _Float16* bRow0 = &Wt[wn * 32 + (lane & 15)][0];
  const _Float16* bRow1 = &Wt[wn * 32 + 16 + (lane & 15)][0];

  for (int k0 = 0; k0 < D_; k0 += 32) {
    {
      const _Float16* src = A + (size_t)(mbase + xr) * D_ + k0 + xc;
      *(uint4*)&Xs[xr][xc] = *(const uint4*)src;   // 8 halves, b128
      if (k0 + 32 < D_) __builtin_prefetch(src + 32, 0, 0);
    }
    {
      const float* src = W + (size_t)(k0 + wr) * D_ + nbase + wc;
      float4 f0 = *(const float4*)(src);
      float4 f1 = *(const float4*)(src + 4);
      if (k0 + 32 < D_) __builtin_prefetch(src + 32 * D_, 0, 0);
      Wt[wc + 0][wr] = (_Float16)f0.x; Wt[wc + 1][wr] = (_Float16)f0.y;
      Wt[wc + 2][wr] = (_Float16)f0.z; Wt[wc + 3][wr] = (_Float16)f0.w;
      Wt[wc + 4][wr] = (_Float16)f1.x; Wt[wc + 5][wr] = (_Float16)f1.y;
      Wt[wc + 6][wr] = (_Float16)f1.z; Wt[wc + 7][wr] = (_Float16)f1.w;
    }
    __syncthreads();
    {
      v16h a  = ldA16(aRow,  lane, 0);
      v16h b0 = ldB16(bRow0, lane, 0);
      v16h b1 = ldB16(bRow1, lane, 0);
      acc0 = wmma_f16(a, b0, acc0);
      acc1 = wmma_f16(a, b1, acc1);
    }
    __syncthreads();
  }

  const int rhi  = (lane >> 4) * 8;
  const int col0 = nbase + wn * 32 + (lane & 15);
  const int col1 = col0 + 16;
  const float b0v = bias[col0];
  const float b1v = bias[col1];
  #pragma unroll
  for (int r = 0; r < 8; ++r) {
    int m = mbase + wm * 16 + r + rhi;
    Y[(size_t)m * D_ + col0] = acc0[r] + b0v;
    Y[(size_t)m * D_ + col1] = acc1[r] + b1v;
  }
}

// ---------------------------------------------------------------------------
extern "C" void kernel_launch(void* const* d_in, const int* in_sizes, int n_in,
                              void* d_out, int out_size, void* d_ws, size_t ws_size,
                              hipStream_t stream) {
  (void)in_sizes; (void)n_in; (void)out_size; (void)ws_size;
  const float* query = (const float*)d_in[0];
  const float* key_  = (const float*)d_in[1];
  const float* value = (const float*)d_in[2];
  const float* pos   = (const float*)d_in[3];
  const float* Wq    = (const float*)d_in[4];
  const float* bq    = (const float*)d_in[5];
  const float* Wk    = (const float*)d_in[6];
  const float* bk    = (const float*)d_in[7];
  const float* Wv    = (const float*)d_in[8];
  const float* bv    = (const float*)d_in[9];
  const float* Wp    = (const float*)d_in[10];
  const float* Wout  = (const float*)d_in[11];
  const float* bout  = (const float*)d_in[12];
  const float* ubias = (const float*)d_in[13];  // [H,DK] flat == per-e vector
  const float* vbias = (const float*)d_in[14];

  char* ws = (char*)d_ws;
  constexpr size_t SZ_QKV = (size_t)B_ * H_ * N_ * DK_ * sizeof(_Float16);  // 8 MB
  constexpr size_t SZ_P   = (size_t)B_ * H_ * P_ * DK_ * sizeof(_Float16);  // ~16 MB
  _Float16* qu_h = (_Float16*)(ws);
  _Float16* qv_h = (_Float16*)(ws + SZ_QKV);
  _Float16* k_h  = (_Float16*)(ws + 2 * SZ_QKV);
  _Float16* vT_h = (_Float16*)(ws + 3 * SZ_QKV);
  _Float16* p_h  = (_Float16*)(ws + 4 * SZ_QKV);
  _Float16* ao_h = (_Float16*)(ws + 4 * SZ_QKV + SZ_P);

  dim3 blk(256);
  // q projection -> qu (q + pos_bias_u) and qv (q + pos_bias_v) in one pass
  rpa_proj_kernel<<<dim3(128, 8), blk, 0, stream>>>(query, Wq, bq, ubias, qu_h, vbias, qv_h, B_ * N_, N_, 0);
  rpa_proj_kernel<<<dim3(128, 8), blk, 0, stream>>>(key_,  Wk, bk, nullptr, k_h,  nullptr, nullptr, B_ * N_, N_, 0);
  rpa_proj_kernel<<<dim3(128, 8), blk, 0, stream>>>(value, Wv, bv, nullptr, vT_h, nullptr, nullptr, B_ * N_, N_, 1);
  rpa_proj_kernel<<<dim3(256, 8), blk, 0, stream>>>(pos,   Wp, nullptr, nullptr, p_h, nullptr, nullptr, B_ * P_, P_, 0);
  rpa_attn_kernel<<<dim3(N_ / 16, H_, B_), blk, 0, stream>>>(qu_h, qv_h, k_h, vT_h, p_h, ao_h);
  rpa_final_kernel<<<dim3(128, 8), blk, 0, stream>>>(ao_h, Wout, bout, (float*)d_out);
}
